// TriAttEnd_old_20194936225862
// MI455X (gfx1250) — matrix operand order repaired and checked by reading, not compiled, not run on hardware
//
#include <hip/hip_runtime.h>
#include <hip/hip_bf16.h>
#include <math.h>

#define L_   192
#define D_   128
#define H_   12
#define C_   24
#define HC_  288
#define LL_  (L_ * L_)

typedef __attribute__((ext_vector_type(16))) __bf16 v16bf;
typedef __attribute__((ext_vector_type(8)))  float  v8f;

union ABop { v16bf v; uint4 q[2]; };

__device__ __forceinline__ v8f wmma_bf16(v16bf a, v16bf b, v8f c) {
  // D = A(16x32 bf16) * B(32x16 bf16) + C(16x16 f32)
  return __builtin_amdgcn_wmma_f32_16x16x32_bf16(false, a, false, b, (short)0, c, false, false);
}

// Branch-free per-dword select (divergent cond -> v_cndmask, no EXEC churn).
__device__ __forceinline__ uint4 sel4(int keep, uint4 a) {
  uint4 r;
  r.x = keep ? a.x : 0u;
  r.y = keep ? a.y : 0u;
  r.z = keep ? a.z : 0u;
  r.w = keep ? a.w : 0u;
  return r;
}

// ---------------------------------------------------------------------------
// Weight transpose + fp32 -> bf16 convert.  dst[Npad, K], dst[n,k] = src[k,n]
// (rows n >= N are zero padding).
// ---------------------------------------------------------------------------
__global__ void transpose_cvt_kernel(const float* __restrict__ src,
                                     __bf16* __restrict__ dst,
                                     int K, int N, int Npad) {
  int idx = blockIdx.x * blockDim.x + threadIdx.x;
  int total = Npad * K;
  if (idx >= total) return;
  int n = idx / K, k = idx - n * K;
  dst[idx] = (n < N) ? (__bf16)src[(size_t)k * N + n] : (__bf16)0.0f;
}

// ---------------------------------------------------------------------------
// LayerNorm over D=128, one wave per row, output bf16.
// ---------------------------------------------------------------------------
__global__ __launch_bounds__(256) void layernorm_kernel(
    const float* __restrict__ z, const float* __restrict__ w,
    const float* __restrict__ b, __bf16* __restrict__ out, int rows) {
  int wave = blockIdx.x * (blockDim.x >> 5) + (threadIdx.x >> 5);
  if (wave >= rows) return;
  int lane = threadIdx.x & 31;
  const float* r = z + (size_t)wave * D_;
  float x[4], s = 0.f, s2 = 0.f;
#pragma unroll
  for (int i = 0; i < 4; ++i) {
    x[i] = r[lane + 32 * i];
    s += x[i];
    s2 += x[i] * x[i];
  }
#pragma unroll
  for (int m = 16; m >= 1; m >>= 1) {
    s  += __shfl_xor(s,  m, 32);
    s2 += __shfl_xor(s2, m, 32);
  }
  float mu  = s * (1.0f / D_);
  float var = s2 * (1.0f / D_) - mu * mu;
  float inv = rsqrtf(var + 1e-5f);
#pragma unroll
  for (int i = 0; i < 4; ++i) {
    int c = lane + 32 * i;
    out[(size_t)wave * D_ + c] = (__bf16)((x[i] - mu) * inv * w[c] + b[c]);
  }
}

// ---------------------------------------------------------------------------
// Register-blocked bf16 WMMA GEMM:
//   out[M, ldOut] = (A[M,K] * BT[Npad,K]^T + bias) * scale
// Each wave computes a (16*MT) x (16*NT) macro-tile: MT+NT operand loads feed
// MT*NT WMMAs per K-step.  K mult of 32; M mult of 16*MT; N mult of 16*NT.
// Stores fp32 (outF) or bf16 (outB); columns >= Nvalid are dropped.
// ---------------------------------------------------------------------------
template <int MT, int NT>
__global__ __launch_bounds__(256) void gemm_bf16_kernel(
    const __bf16* __restrict__ A, const __bf16* __restrict__ BT,
    const float* __restrict__ bias, float* __restrict__ outF,
    __bf16* __restrict__ outB, int M, int N, int K, int Nvalid, int ldOut,
    float scale) {
  int wave = blockIdx.x * (blockDim.x >> 5) + (threadIdx.x >> 5);
  int nG = N / (16 * NT);
  int total = (M / (16 * MT)) * nG;
  if (wave >= total) return;
  int m0 = (wave / nG) * 16 * MT;
  int n0 = (wave % nG) * 16 * NT;
  int lane = threadIdx.x & 31;
  int half = lane >> 4, lr = lane & 15;
  int kbA = half << 3;    // A: K in {kbA..kbA+7, kbA+16..kbA+23}
  int kbB = half << 3;    // B: K in {kbB..kbB+7, kbB+16..kbB+23}

  const __bf16* arow[MT];
  const __bf16* brow[NT];
#pragma unroll
  for (int i = 0; i < MT; ++i) arow[i] = A  + (size_t)(m0 + i * 16 + lr) * K;
#pragma unroll
  for (int j = 0; j < NT; ++j) brow[j] = BT + (size_t)(n0 + j * 16 + lr) * K;

  const v8f vzero = {0, 0, 0, 0, 0, 0, 0, 0};
  v8f acc[MT][NT];
#pragma unroll
  for (int i = 0; i < MT; ++i)
#pragma unroll
    for (int j = 0; j < NT; ++j) acc[i][j] = vzero;

  for (int k0 = 0; k0 < K; k0 += 32) {
    ABop a[MT], b[NT];
#pragma unroll
    for (int i = 0; i < MT; ++i) {
      a[i].q[0] = *(const uint4*)(arow[i] + k0 + kbA);
      a[i].q[1] = *(const uint4*)(arow[i] + k0 + kbA + 16);
    }
#pragma unroll
    for (int j = 0; j < NT; ++j) {
      b[j].q[0] = *(const uint4*)(brow[j] + k0 + kbB);
      b[j].q[1] = *(const uint4*)(brow[j] + k0 + kbB + 16);
    }
#pragma unroll
    for (int i = 0; i < MT; ++i)
#pragma unroll
      for (int j = 0; j < NT; ++j)
        acc[i][j] = wmma_bf16(a[i].v, b[j].v, acc[i][j]);
  }

#pragma unroll
  for (int i = 0; i < MT; ++i) {
#pragma unroll
    for (int j = 0; j < NT; ++j) {
      int n = n0 + j * 16 + lr;
      if (n < Nvalid) {
        float bv = bias ? bias[n] : 0.0f;
#pragma unroll
        for (int r = 0; r < 8; ++r) {
          int m = m0 + i * 16 + r + (half << 3);
          float v = (acc[i][j][r] + bv) * scale;
          if (outF) outF[(size_t)m * ldOut + n] = v;
          else      outB[(size_t)m * ldOut + n] = (__bf16)v;
        }
      }
    }
  }
}

// ---------------------------------------------------------------------------
// Pass 1: scores + softmax.  One wave per (b, h, l-tile).
//   S[l,k] = sum_c q[b,l,h,c] * k[k,b,h,c]   (q pre-scaled by 1/sqrt(C))
//   S += bias[k,l,h];  softmax over k (192);  att bf16 -> [l,h,b,k]
// All operand loads are unconditional (addresses always in-bounds inside the
// 288-wide row); C=24 -> K=32 padding done with branch-free v_cndmask zeroing.
// ---------------------------------------------------------------------------
__global__ __launch_bounds__(256) void attn_scores_kernel(
    const __bf16* __restrict__ qb, const __bf16* __restrict__ kb,
    const float* __restrict__ biasb, __bf16* __restrict__ attb) {
  const int LT = L_ / 16;  // 12
  int wave = blockIdx.x * (blockDim.x >> 5) + (threadIdx.x >> 5);
  int lt = wave % LT;
  int tmp = wave / LT;
  int h = tmp % H_;
  int b = tmp / H_;
  if (b >= L_) return;
  int lane = threadIdx.x & 31;
  int half = lane >> 4, lr = lane & 15;
  int keep = (half == 0);           // lanes 0-15 keep the second chunk

  // A operand: Q rows (l = lt*16 + lr), c chunks {8h..8h+7, 16..23|pad}.
  ABop a;
  const __bf16* qrow = qb + ((size_t)(b * L_ + lt * 16 + lr)) * HC_ + h * C_;
  a.q[0] = *(const uint4*)(qrow + (half << 3));           // c 0..7 | 8..15
  a.q[1] = sel4(keep, *(const uint4*)(qrow + 16));        // c 16..23 | zero

  v8f acc[12];
#pragma unroll
  for (int t = 0; t < 12; ++t) {
    // B operand: column n = t*16+lr is K2 row n = k_[n,b,h,:].
    // K chunks {8h..8h+7, 16..23|pad}: q0 at +8h (always valid), q1 = +16 | zero.
    ABop bb;
    const __bf16* krow =
        kb + ((size_t)((t * 16 + lr) * L_ + b)) * HC_ + h * C_;
    bb.q[0] = *(const uint4*)(krow + (half << 3));        // c 0..7 | 8..15
    bb.q[1] = sel4(keep, *(const uint4*)(krow + 16));     // c 16..23 | zero
    v8f c0 = {0, 0, 0, 0, 0, 0, 0, 0};
    acc[t] = wmma_bf16(a.v, bb.v, c0);
  }

  // bias add + softmax over the 192 keys (per row: 12 tiles x 16 lanes of half)
#pragma unroll
  for (int i = 0; i < 8; ++i) {
    int l = lt * 16 + i + (half << 3);
    float mx = -3.4e38f;
#pragma unroll
    for (int t = 0; t < 12; ++t) {
      int kc = t * 16 + lr;
      acc[t][i] += biasb[((size_t)kc * L_ + l) * H_ + h];
      mx = fmaxf(mx, acc[t][i]);
    }
#pragma unroll
    for (int m = 8; m >= 1; m >>= 1) mx = fmaxf(mx, __shfl_xor(mx, m, 32));
    float sum = 0.f;
#pragma unroll
    for (int t = 0; t < 12; ++t) {
      acc[t][i] = __expf(acc[t][i] - mx);
      sum += acc[t][i];
    }
#pragma unroll
    for (int m = 8; m >= 1; m >>= 1) sum += __shfl_xor(sum, m, 32);
    float r = 1.0f / sum;
#pragma unroll
    for (int t = 0; t < 12; ++t) {
      int kc = t * 16 + lr;
      attb[(((size_t)l * H_ + h) * L_ + b) * L_ + kc] = (__bf16)(acc[t][i] * r);
    }
  }
}

// ---------------------------------------------------------------------------
// Pass 2: o[b,l,h,c] = sum_k att[l,h,b,k] * v[k,l,h,c]; fused sigmoid gating.
// One block per (l,h); V_lh staged transposed into LDS (C padded 24->32).
// ---------------------------------------------------------------------------
__global__ __launch_bounds__(256) void attn_av_kernel(
    const __bf16* __restrict__ attb, const __bf16* __restrict__ vb,
    const __bf16* __restrict__ gb, __bf16* __restrict__ ogb) {
  __shared__ __bf16 vt[32 * L_];  // vt[c][k], 12 KB
  int l = blockIdx.x / H_;
  int h = blockIdx.x % H_;
  int tid = threadIdx.x;
  for (int idx = tid; idx < L_ * 32; idx += 256) {
    int k = idx >> 5, c = idx & 31;
    __bf16 val = (__bf16)0.0f;
    if (c < C_) val = vb[((size_t)(k * L_ + l)) * HC_ + h * C_ + c];
    vt[c * L_ + k] = val;
  }
  __syncthreads();

  int wv = tid >> 5, lane = tid & 31;
  int half = lane >> 4, lr = lane & 15;
  int kbA = half << 3, kbB = half << 3;
  for (int t = wv; t < 24; t += 8) {   // 12 m-tiles (b) x 2 n-tiles (c)
    int mt = t >> 1, nt = t & 1;
    const __bf16* arow =
        attb + (((size_t)l * H_ + h) * L_ + mt * 16 + lr) * L_;
    const __bf16* brow = vt + (nt * 16 + lr) * L_;
    v8f acc = {0, 0, 0, 0, 0, 0, 0, 0};
#pragma unroll
    for (int k0 = 0; k0 < L_; k0 += 32) {
      ABop a, b;
      a.q[0] = *(const uint4*)(arow + k0 + kbA);
      a.q[1] = *(const uint4*)(arow + k0 + kbA + 16);
      b.q[0] = *(const uint4*)(brow + k0 + kbB);       // LDS ds_read_b128
      b.q[1] = *(const uint4*)(brow + k0 + kbB + 16);
      acc = wmma_bf16(a.v, b.v, acc);
    }
    int c = nt * 16 + lr;
    if (c < C_) {
#pragma unroll
      for (int i = 0; i < 8; ++i) {
        int bi = mt * 16 + i + (half << 3);
        size_t off = ((size_t)(bi * L_ + l)) * HC_ + h * C_ + c;
        float gv = (float)gb[off];
        float gate = 1.0f / (1.0f + __expf(-gv));
        ogb[off] = (__bf16)(gate * acc[i]);
      }
    }
  }
}

// ---------------------------------------------------------------------------
extern "C" void kernel_launch(void* const* d_in, const int* in_sizes, int n_in,
                              void* d_out, int out_size, void* d_ws,
                              size_t ws_size, hipStream_t stream) {
  const float* z_   = (const float*)d_in[0];
  const float* ln_w = (const float*)d_in[1];
  const float* ln_b = (const float*)d_in[2];
  const float* Wq = (const float*)d_in[3];
  const float* bq = (const float*)d_in[4];
  const float* Wk = (const float*)d_in[5];
  const float* bk = (const float*)d_in[6];
  const float* Wv = (const float*)d_in[7];
  const float* bv = (const float*)d_in[8];
  const float* Wb = (const float*)d_in[9];
  const float* bb = (const float*)d_in[10];
  const float* Wg = (const float*)d_in[11];
  const float* bg = (const float*)d_in[12];
  const float* Wo = (const float*)d_in[13];
  const float* bo = (const float*)d_in[14];
  float* out = (float*)d_out;

  // ---- workspace carve (256-byte aligned) ----
  char* ws = (char*)d_ws;
  auto carve = [&](size_t bytes) -> void* {
    void* p = (void*)ws;
    ws += (bytes + 255) & ~(size_t)255;
    return p;
  };
  __bf16* zbf  = (__bf16*)carve((size_t)LL_ * D_ * 2);
  __bf16* WqT  = (__bf16*)carve((size_t)HC_ * D_ * 2);
  __bf16* WkT  = (__bf16*)carve((size_t)HC_ * D_ * 2);
  __bf16* WvT  = (__bf16*)carve((size_t)HC_ * D_ * 2);
  __bf16* WgT  = (__bf16*)carve((size_t)HC_ * D_ * 2);
  __bf16* WbT  = (__bf16*)carve((size_t)16 * D_ * 2);
  __bf16* WoT  = (__bf16*)carve((size_t)D_ * HC_ * 2);
  __bf16* qbuf = (__bf16*)carve((size_t)LL_ * HC_ * 2);
  __bf16* kbuf = (__bf16*)carve((size_t)LL_ * HC_ * 2);
  __bf16* vbuf = (__bf16*)carve((size_t)LL_ * HC_ * 2);
  __bf16* gbuf = (__bf16*)carve((size_t)LL_ * HC_ * 2);
  float*  bias = (float*)carve((size_t)LL_ * H_ * 4);
  __bf16* attb = (__bf16*)carve((size_t)L_ * H_ * L_ * L_ * 2);  // 170 MB
  __bf16* ogb  = (__bf16*)carve((size_t)LL_ * HC_ * 2);

  auto cdiv = [](int a, int b) { return (a + b - 1) / b; };
  const float sq_c = 1.0f / sqrtf((float)C_);

  // 1) weight transpose + bf16 convert
  int wTot = HC_ * D_;
  transpose_cvt_kernel<<<cdiv(wTot, 256), 256, 0, stream>>>(Wq, WqT, D_, HC_, HC_);
  transpose_cvt_kernel<<<cdiv(wTot, 256), 256, 0, stream>>>(Wk, WkT, D_, HC_, HC_);
  transpose_cvt_kernel<<<cdiv(wTot, 256), 256, 0, stream>>>(Wv, WvT, D_, HC_, HC_);
  transpose_cvt_kernel<<<cdiv(wTot, 256), 256, 0, stream>>>(Wg, WgT, D_, HC_, HC_);
  transpose_cvt_kernel<<<cdiv(16 * D_, 256), 256, 0, stream>>>(Wb, WbT, D_, H_, 16);
  transpose_cvt_kernel<<<cdiv(D_ * HC_, 256), 256, 0, stream>>>(Wo, WoT, HC_, D_, D_);

  // 2) LayerNorm -> bf16
  layernorm_kernel<<<cdiv(LL_, 8), 256, 0, stream>>>(z_, ln_w, ln_b, zbf, LL_);

  // 3) projections (q pre-scaled by 1/sqrt(C), bias folded before scale)
  // 32x32 macro-tiles: waves = (M/32)*(N/32)
  int wavesP = (LL_ / 32) * (HC_ / 32);  // 1152 * 9
  gemm_bf16_kernel<2, 2><<<cdiv(wavesP, 8), 256, 0, stream>>>(
      zbf, WqT, bq, nullptr, qbuf, LL_, HC_, D_, HC_, HC_, sq_c);
  gemm_bf16_kernel<2, 2><<<cdiv(wavesP, 8), 256, 0, stream>>>(
      zbf, WkT, bk, nullptr, kbuf, LL_, HC_, D_, HC_, HC_, 1.0f);
  gemm_bf16_kernel<2, 2><<<cdiv(wavesP, 8), 256, 0, stream>>>(
      zbf, WvT, bv, nullptr, vbuf, LL_, HC_, D_, HC_, HC_, 1.0f);
  gemm_bf16_kernel<2, 2><<<cdiv(wavesP, 8), 256, 0, stream>>>(
      zbf, WgT, bg, nullptr, gbuf, LL_, HC_, D_, HC_, HC_, 1.0f);
  int wavesB = LL_ / 32;  // 32x16 macro-tiles, N padded to 16
  gemm_bf16_kernel<2, 1><<<cdiv(wavesB, 8), 256, 0, stream>>>(
      zbf, WbT, bb, bias, nullptr, LL_, 16, D_, H_, H_, 1.0f);

  // 4) scores + softmax
  int wavesS = L_ * H_ * (L_ / 16);  // 27648
  attn_scores_kernel<<<cdiv(wavesS, 8), 256, 0, stream>>>(qbuf, kbuf, bias, attb);

  // 5) att @ V with fused sigmoid gating
  attn_av_kernel<<<L_ * H_, 256, 0, stream>>>(attb, vbuf, gbuf, ogb);

  // 6) output projection -> fp32 out
  int wavesO = (LL_ / 32) * (D_ / 32);  // 1152 * 4
  gemm_bf16_kernel<2, 2><<<cdiv(wavesO, 8), 256, 0, stream>>>(
      ogb, WoT, bo, out, nullptr, LL_, D_, HC_, D_, D_, 1.0f);
}
